// DLRM_Net_15229954032043
// MI455X (gfx1250) — matrix operand order repaired and checked by reading, not compiled, not run on hardware
//
#include <hip/hip_runtime.h>
#include <hip/hip_bf16.h>
#include <stdint.h>

typedef __attribute__((ext_vector_type(16))) __bf16 v16bf;
typedef __attribute__((ext_vector_type(8)))  float  v8f;

constexpr int BATCH = 32768;
constexpr int NTAB  = 26;
constexpr int NROWS = 200000;
constexpr int DIM   = 64;
constexpr int RW    = 448;     // padded R width (415 real cols, pad -> mult of 64)

union FragBF { v16bf v; uint4 u[2]; };

#if defined(__has_builtin)
#if __has_builtin(__builtin_amdgcn_sched_group_barrier)
#define HAVE_SGB 1
#endif
#endif
#ifdef HAVE_SGB
// pattern per unrolled body: 10 vmem reads, 4 wmma, 10 vmem reads, 4 wmma
#define SCHED_PIPELINE()                                   \
  do {                                                     \
    __builtin_amdgcn_sched_group_barrier(0x020, 10, 0);    \
    __builtin_amdgcn_sched_group_barrier(0x008, 4, 0);     \
    __builtin_amdgcn_sched_group_barrier(0x020, 10, 0);    \
    __builtin_amdgcn_sched_group_barrier(0x008, 4, 0);     \
  } while (0)
#else
#define SCHED_PIPELINE()
#endif

// ---------------------------------------------------------------------------
// Pack dense_x (B x 13 fp32) -> Xb (B x 64 bf16, zero padded K)
// ---------------------------------------------------------------------------
__global__ __launch_bounds__(256)
void pack_dense_kernel(const float* __restrict__ x, __bf16* __restrict__ xb) {
  int tid = blockIdx.x * 256 + threadIdx.x;      // BATCH*64 threads
  int b = tid >> 6, k = tid & 63;
  float v = (k < 13) ? x[b * 13 + k] : 0.0f;
  xb[tid] = (__bf16)v;
}

// ---------------------------------------------------------------------------
// Pack weight (N x K_actual fp32 row-major) -> (N x K_pad bf16, zero padded)
// ---------------------------------------------------------------------------
__global__ __launch_bounds__(256)
void pack_weight_kernel(const float* __restrict__ W, __bf16* __restrict__ Wb,
                        int K_actual, int K_pad, int total) {
  int tid = blockIdx.x * 256 + threadIdx.x;
  if (tid >= total) return;
  int n = tid / K_pad, k = tid - n * K_pad;
  Wb[tid] = (__bf16)((k < K_actual) ? W[(size_t)n * K_actual + k] : 0.0f);
}

// ---------------------------------------------------------------------------
// GEMM fragment loaders (row-major bf16, per-lane contiguous 16B chunks)
// ---------------------------------------------------------------------------
__device__ __forceinline__ void load_afrag(const __bf16* arow, int k0, int half,
                                           FragBF& af) {
  af.u[0] = *(const uint4*)(arow + k0 + half * 16);
  af.u[1] = *(const uint4*)(arow + k0 + half * 16 + 8);
}
__device__ __forceinline__ void load_bfrag(const __bf16* const wrow[4], int k0,
                                           FragBF bf[4]) {
  #pragma unroll
  for (int nt = 0; nt < 4; ++nt) {
    bf[nt].u[0] = *(const uint4*)(wrow[nt] + k0);
    bf[nt].u[1] = *(const uint4*)(wrow[nt] + k0 + 8);
  }
}

// ---------------------------------------------------------------------------
// C[m][n] = relu( sum_k A[m][k] * Wb[n][k] + bias[n] ),  all operands bf16,
// f32 accumulate.  K % 64 == 0, rows 16B aligned.  No LDS.  2-deep software
// pipeline: prefetch next K-step fragments under the current 4 WMMAs.
// NOTE: prefetch over-reads <=128B past the operand tails on the last
// iteration (never consumed) -- all operands are ws buffers padded +256B.
// Block tile 128(M) x 64(N): 8 waves, one 16-row strip per wave, 4 N-tiles.
// ---------------------------------------------------------------------------
__global__ __launch_bounds__(256)
void gemm_bias_relu(const __bf16* __restrict__ A, int lda,
                    const __bf16* __restrict__ Wb, int ldw,
                    const float* __restrict__ bias,
                    __bf16* __restrict__ C, int ldc, int K)
{
  const int lane = threadIdx.x & 31;
  const int wv   = threadIdx.x >> 5;
  const int half = lane >> 4;
  const int l16  = lane & 15;

  const int m  = blockIdx.x * 128 + wv * 16 + l16;  // A row this lane reads
  const int n0 = blockIdx.y * 64;

  const __bf16* arow = A + (size_t)m * lda;
  const __bf16* wrow[4];
  #pragma unroll
  for (int nt = 0; nt < 4; ++nt)
    wrow[nt] = Wb + (size_t)(n0 + nt * 16 + l16) * ldw + half * 16;

  v8f acc[4];
  #pragma unroll
  for (int t = 0; t < 4; ++t)
    #pragma unroll
    for (int i = 0; i < 8; ++i) acc[t][i] = 0.0f;

  FragBF afA, afB, bfA[4], bfB[4];
  load_afrag(arow, 0, half, afA);
  load_bfrag(wrow, 0, bfA);

  for (int k0 = 0; k0 < K; k0 += 64) {
    // phase 0: prefetch k0+32, compute on k0
    load_afrag(arow, k0 + 32, half, afB);
    load_bfrag(wrow, k0 + 32, bfB);
    #pragma unroll
    for (int nt = 0; nt < 4; ++nt)
      acc[nt] = __builtin_amdgcn_wmma_f32_16x16x32_bf16(
          false, afA.v, false, bfA[nt].v, (short)0, acc[nt], false, false);

    // phase 1: prefetch k0+64 (may over-read on last iter), compute on k0+32
    load_afrag(arow, k0 + 64, half, afA);
    load_bfrag(wrow, k0 + 64, bfA);
    #pragma unroll
    for (int nt = 0; nt < 4; ++nt)
      acc[nt] = __builtin_amdgcn_wmma_f32_16x16x32_bf16(
          false, afB.v, false, bfB[nt].v, (short)0, acc[nt], false, false);

    SCHED_PIPELINE();
  }

  // epilogue: bias + relu, store bf16. D layout: VGPR r -> M = r + 8*half.
  #pragma unroll
  for (int nt = 0; nt < 4; ++nt) {
    #pragma unroll
    for (int r = 0; r < 8; ++r) {
      int mm = blockIdx.x * 128 + wv * 16 + r + half * 8;
      int nn = n0 + nt * 16 + l16;
      float v = acc[nt][r] + bias[nn];
      v = v > 0.0f ? v : 0.0f;
      C[(size_t)mm * ldc + nn] = (__bf16)v;
    }
  }
}

// ---------------------------------------------------------------------------
// Interaction: one wave per batch row. T = [x_bot; emb_0..emb_25] (27x64,
// padded to 32x64 in LDS). Z = T T^T via WMMA, lower-triangle tiles only;
// scatter strict lower triangle into R[:, 64:415], zero pad cols 415:448.
// ---------------------------------------------------------------------------
__global__ __launch_bounds__(256)
void interact_kernel(const int* __restrict__ indices,
                     const float* __restrict__ emb,
                     __bf16* __restrict__ R)
{
  __shared__ __bf16 sT[8][32][72];   // 72-elem stride -> conflict-free frag reads
  const int lane = threadIdx.x & 31;
  const int wv   = threadIdx.x >> 5;
  const int half = lane >> 4;
  const int l16  = lane & 15;
  const int b    = blockIdx.x * 8 + wv;

  // row 0: x_bot already bf16 in R[:, 0:64]
  {
    const __bf16* src = R + (size_t)b * RW;
    sT[wv][0][lane * 2]     = src[lane * 2];
    sT[wv][0][lane * 2 + 1] = src[lane * 2 + 1];
  }
  // rows 1..26: gather embeddings (coalesced 256B per row), fp32 -> bf16
  for (int t = 0; t < NTAB; ++t) {
    int idx = indices[(size_t)t * BATCH + b];
    const float2* src = (const float2*)(emb + ((size_t)t * NROWS + idx) * DIM);
    float2 v = src[lane];
    sT[wv][1 + t][lane * 2]     = (__bf16)v.x;
    sT[wv][1 + t][lane * 2 + 1] = (__bf16)v.y;
  }
  // rows 27..31: zero pad
  #pragma unroll
  for (int r = 27; r < 32; ++r) {
    sT[wv][r][lane * 2]     = (__bf16)0.0f;
    sT[wv][r][lane * 2 + 1] = (__bf16)0.0f;
  }

  // only lower-triangle tiles of the 32x32 output: (0,0) (1,0) (1,1)
  const int tm[3] = {0, 1, 1};
  const int tn[3] = {0, 0, 1};
  __bf16* rrow = R + (size_t)b * RW + DIM;

  #pragma unroll
  for (int tI = 0; tI < 3; ++tI) {
    const int mt = tm[tI], nt = tn[tI];
    v8f acc;
    #pragma unroll
    for (int i = 0; i < 8; ++i) acc[i] = 0.0f;

    #pragma unroll
    for (int k0 = 0; k0 < 64; k0 += 32) {
      v16bf af, bfv;
      const __bf16* pa = &sT[wv][mt * 16 + l16][k0];
      #pragma unroll
      for (int v = 0; v < 4; ++v) {
        int k = half * 8 + 2 * v;
        af[2 * v]         = pa[k];
        af[2 * v + 1]     = pa[k + 1];
        af[8 + 2 * v]     = pa[16 + k];
        af[8 + 2 * v + 1] = pa[16 + k + 1];
      }
      const __bf16* pb = &sT[wv][nt * 16 + l16][k0];   // B[k][n] = T[n][k]
      #pragma unroll
      for (int v = 0; v < 8; ++v) {
        int k = half * 8 + 2 * v + (v >= 4 ? 8 : 0);
        bfv[2 * v]     = pb[k];
        bfv[2 * v + 1] = pb[k + 1];
      }
      acc = __builtin_amdgcn_wmma_f32_16x16x32_bf16(
          false, af, false, bfv, (short)0, acc, false, false);
    }

    // scatter strict lower triangle: flat index i*(i-1)/2 + j (tril k=-1 order)
    #pragma unroll
    for (int r = 0; r < 8; ++r) {
      int i = mt * 16 + r + half * 8;
      int j = nt * 16 + l16;
      if (i >= 1 && i <= NTAB && j < i)
        rrow[i * (i - 1) / 2 + j] = (__bf16)acc[r];
    }
  }
  // zero pad cols 415..447
  for (int c = 415 + lane; c < RW; c += 32)
    R[(size_t)b * RW + c] = (__bf16)0.0f;
}

// ---------------------------------------------------------------------------
// Final layer: p[b] = sigmoid( dot(h1[b], tw2) + tb2 ). One wave per row.
// ---------------------------------------------------------------------------
__global__ __launch_bounds__(256)
void final_kernel(const __bf16* __restrict__ h, const float* __restrict__ w,
                  const float* __restrict__ bias, float* __restrict__ out)
{
  const int lane = threadIdx.x & 31;
  const int wv   = threadIdx.x >> 5;
  const int b    = blockIdx.x * 8 + wv;

  uint4 raw = *(const uint4*)(h + (size_t)b * 256 + lane * 8);
  const __bf16* e = (const __bf16*)&raw;
  float s = 0.0f;
  #pragma unroll
  for (int i = 0; i < 8; ++i) s += (float)e[i] * w[lane * 8 + i];
  #pragma unroll
  for (int off = 16; off > 0; off >>= 1) s += __shfl_xor(s, off, 32);
  if (lane == 0) out[b] = 1.0f / (1.0f + __expf(-(s + bias[0])));
}

// ---------------------------------------------------------------------------
extern "C" void kernel_launch(void* const* d_in, const int* in_sizes, int n_in,
                              void* d_out, int out_size, void* d_ws, size_t ws_size,
                              hipStream_t stream)
{
  (void)in_sizes; (void)n_in; (void)out_size; (void)ws_size;
  const float* dense_x = (const float*)d_in[0];
  const int*   indices = (const int*)d_in[1];     // harness contract: integer -> const int*
  const float* emb     = (const float*)d_in[2];
  const float* bw0 = (const float*)d_in[3];
  const float* bb0 = (const float*)d_in[4];
  const float* bw1 = (const float*)d_in[5];
  const float* bb1 = (const float*)d_in[6];
  const float* bw2 = (const float*)d_in[7];
  const float* bb2 = (const float*)d_in[8];
  const float* tw0 = (const float*)d_in[9];
  const float* tb0 = (const float*)d_in[10];
  const float* tw1 = (const float*)d_in[11];
  const float* tb1 = (const float*)d_in[12];
  const float* tw2 = (const float*)d_in[13];
  const float* tb2 = (const float*)d_in[14];
  float* out = (float*)d_out;

  char* ws = (char*)d_ws;
  size_t off = 0;
  auto alloc = [&](size_t bytes) {
    char* p = ws + off;
    off = (off + bytes + 256 + 255) & ~(size_t)255;   // +256B tail pad (prefetch over-read)
    return p;
  };
  __bf16* Xb   = (__bf16*)alloc((size_t)BATCH * 64  * 2);
  __bf16* H0   = (__bf16*)alloc((size_t)BATCH * 512 * 2);
  __bf16* H1   = (__bf16*)alloc((size_t)BATCH * 256 * 2);
  __bf16* Rb   = (__bf16*)alloc((size_t)BATCH * RW  * 2);
  __bf16* W0b  = (__bf16*)alloc((size_t)512 * 64  * 2);
  __bf16* W1b  = (__bf16*)alloc((size_t)256 * 512 * 2);
  __bf16* W2b  = (__bf16*)alloc((size_t)64  * 256 * 2);
  __bf16* TW0b = (__bf16*)alloc((size_t)512 * RW  * 2);
  __bf16* TW1b = (__bf16*)alloc((size_t)256 * 512 * 2);
  __bf16* T0 = H0;   // reuse: H0 fully consumed before T0 written
  __bf16* T1 = H1;   // reuse: H1 fully consumed before T1 written

  // one-shot packs (tiny)
  pack_dense_kernel<<<BATCH * 64 / 256, 256, 0, stream>>>(dense_x, Xb);
  pack_weight_kernel<<<(512 * 64  + 255) / 256, 256, 0, stream>>>(bw0, W0b, 13,  64,  512 * 64);
  pack_weight_kernel<<<(256 * 512 + 255) / 256, 256, 0, stream>>>(bw1, W1b, 512, 512, 256 * 512);
  pack_weight_kernel<<<(64  * 256 + 255) / 256, 256, 0, stream>>>(bw2, W2b, 256, 256, 64 * 256);
  pack_weight_kernel<<<(512 * RW  + 255) / 256, 256, 0, stream>>>(tw0, TW0b, 415, RW, 512 * RW);
  pack_weight_kernel<<<(256 * 512 + 255) / 256, 256, 0, stream>>>(tw1, TW1b, 512, 512, 256 * 512);

  // bottom MLP
  gemm_bias_relu<<<dim3(BATCH / 128, 512 / 64), 256, 0, stream>>>(
      Xb, 64, W0b, 64, bb0, H0, 512, 64);
  gemm_bias_relu<<<dim3(BATCH / 128, 256 / 64), 256, 0, stream>>>(
      H0, 512, W1b, 512, bb1, H1, 256, 512);
  gemm_bias_relu<<<dim3(BATCH / 128, 64 / 64), 256, 0, stream>>>(
      H1, 256, W2b, 256, bb2, Rb, RW, 256);       // x_bot -> R[:, 0:64]

  // pairwise interaction -> R[:, 64:415] (+ zero pad cols)
  interact_kernel<<<BATCH / 8, 256, 0, stream>>>(indices, emb, Rb);

  // top MLP
  gemm_bias_relu<<<dim3(BATCH / 128, 512 / 64), 256, 0, stream>>>(
      Rb, RW, TW0b, RW, tb0, T0, 512, RW);
  gemm_bias_relu<<<dim3(BATCH / 128, 256 / 64), 256, 0, stream>>>(
      T0, 512, TW1b, 512, tb1, T1, 256, 512);
  final_kernel<<<BATCH / 8, 256, 0, stream>>>(T1, tw2, tb2, out);
}
